// Attention_63084479643842
// MI455X (gfx1250) — compile-verified, matrix operand order checked by reading, not run on hardware
//
#include <hip/hip_runtime.h>
#include <hip/hip_bf16.h>

// ---------------------------------------------------------------------------
// Shapes (fixed by the reference):
//   B=4, S=32, Kseq=32  -> T = 1024 tokens/batch, TOK = 4096
//   D=4096, H=32 heads, KV=8 kv-heads, HD=128, GQA rep = 4
// ---------------------------------------------------------------------------
#define TOK   4096
#define TSEQ  1024
#define DMODEL 4096
#define NH    32
#define NKV   8
#define HD    128

typedef __attribute__((ext_vector_type(16))) __bf16   v16bf;
typedef __attribute__((ext_vector_type(8)))  float    v8f;
typedef __attribute__((ext_vector_type(4)))  unsigned int u32x4;

struct Frag32B { u32x4 lo, hi; };

// Load a 16x32 bf16 A/B fragment slice for this lane.
// Caller passes p = row_base + kbase + 8*g  (g = lane>>4). Per ISA 7.12.2:
// VGPR0-3 hold K = kbase+8g .. +7, VGPR4-7 hold K = kbase+16+8g .. +7.
__device__ inline v16bf load_frag16(const __hip_bfloat16* p) {
    Frag32B t;
    t.lo = *reinterpret_cast<const u32x4*>(p);
    t.hi = *reinterpret_cast<const u32x4*>(p + 16);
    return __builtin_bit_cast(v16bf, t);
}

__device__ inline v16bf ones_frag() {
    u32x4 o4 = (u32x4)(0x3f803f80u);      // two packed bf16 1.0
    Frag32B t;
    t.lo = o4; t.hi = o4;
    return __builtin_bit_cast(v16bf, t);
}

__device__ inline v8f wmma_bf16(v16bf a, v16bf b, v8f c) {
    return __builtin_amdgcn_wmma_f32_16x16x32_bf16(
        /*neg_a=*/false, a, /*neg_b=*/false, b,
        /*c_mod=*/(short)0, c, /*reuse_a=*/false, /*reuse_b=*/false);
}

// Max across each 16-lane group using immediate ds_swizzle (SWAPX1/2/4/8),
// avoiding the ds_bpermute + index-VGPR path that __shfl_xor produces.
__device__ inline float swz_max16(float v) {
    v = fmaxf(v, __uint_as_float((unsigned)__builtin_amdgcn_ds_swizzle(
            (int)__float_as_uint(v), 0x041f)));                     // xor 1
    v = fmaxf(v, __uint_as_float((unsigned)__builtin_amdgcn_ds_swizzle(
            (int)__float_as_uint(v), 0x081f)));                     // xor 2
    v = fmaxf(v, __uint_as_float((unsigned)__builtin_amdgcn_ds_swizzle(
            (int)__float_as_uint(v), 0x101f)));                     // xor 4
    v = fmaxf(v, __uint_as_float((unsigned)__builtin_amdgcn_ds_swizzle(
            (int)__float_as_uint(v), 0x201f)));                     // xor 8
    return v;
}

// ---------------------------------------------------------------------------
// fp32 -> bf16 conversion
// ---------------------------------------------------------------------------
__global__ void __launch_bounds__(256)
cvt_f32_bf16(const float* __restrict__ in, __hip_bfloat16* __restrict__ out, long n) {
    long i = (long)blockIdx.x * blockDim.x + threadIdx.x;
    if (i < n) out[i] = __float2bfloat16(in[i]);
}

// ---------------------------------------------------------------------------
// GEMM: C[M,N] = A[M,K] (bf16 row-major) * B[N,K]^T (bf16, rows=N)
// Wave tile 32x64 (2 A-frags x 4 B-frags -> 8 WMMAs / k-step).
// Block = 128 threads (4 waves) -> 128x64. Grid = (M/128, N/64).
// ---------------------------------------------------------------------------
template <typename OUT>
__device__ inline void store_out(OUT* p, float v);
template <> __device__ inline void store_out<float>(float* p, float v) { *p = v; }
template <> __device__ inline void store_out<__hip_bfloat16>(__hip_bfloat16* p, float v) {
    *p = __float2bfloat16(v);
}

template <typename OUT>
__global__ void __launch_bounds__(128, 1)
gemm_bf16_nt(const __hip_bfloat16* __restrict__ A,
             const __hip_bfloat16* __restrict__ B,
             OUT* __restrict__ C, int M, int N, int K) {
    const int wave  = threadIdx.x >> 5;
    const int lane  = threadIdx.x & 31;
    const int m_    = lane & 15;
    const int g     = lane >> 4;
    const int mbase = blockIdx.x * 128 + wave * 32;
    const int nbase = blockIdx.y * 64;

    v8f acc0[4] = {};
    v8f acc1[4] = {};

    const __hip_bfloat16* arow0 = A + (size_t)(mbase + m_) * K + 8 * g;
    const __hip_bfloat16* arow1 = A + (size_t)(mbase + 16 + m_) * K + 8 * g;
    const __hip_bfloat16* brow  = B + (size_t)(nbase + m_) * K + 8 * g;

    for (int k0 = 0; k0 < K; k0 += 32) {
        if (k0 + 32 < K) {
            __builtin_prefetch(arow0 + k0 + 32, 0, 0);          // global_prefetch_b8
            __builtin_prefetch(arow1 + k0 + 32, 0, 0);
        }
        v16bf a0 = load_frag16(arow0 + k0);
        v16bf a1 = load_frag16(arow1 + k0);
#pragma unroll
        for (int nt = 0; nt < 4; ++nt) {
            v16bf b = load_frag16(brow + (size_t)(nt * 16) * K + k0);
            acc0[nt] = wmma_bf16(a0, b, acc0[nt]);
            acc1[nt] = wmma_bf16(a1, b, acc1[nt]);
        }
    }

#pragma unroll
    for (int nt = 0; nt < 4; ++nt) {
#pragma unroll
        for (int j = 0; j < 8; ++j) {
            int r = mbase + j + 8 * g;       // C layout: VGPR j -> row j + 8*(lane/16)
            int c = nbase + nt * 16 + m_;    //            col = lane % 16
            store_out<OUT>(C + (size_t)r * N + c, acc0[nt][j]);
            store_out<OUT>(C + (size_t)(r + 16) * N + c, acc1[nt][j]);
        }
    }
}

// ---------------------------------------------------------------------------
// RoPE in place on bf16 activations laid out [token][heads*HD].
// `oscale` folds the attention 1/sqrt(HD) into Q here (1.0 for K).
// ---------------------------------------------------------------------------
__global__ void __launch_bounds__(256)
rope_kernel(__hip_bfloat16* __restrict__ buf, const float* __restrict__ freqs,
            int heads, float oscale) {
    long idx = (long)blockIdx.x * blockDim.x + threadIdx.x;
    long total = (long)TOK * heads * (HD / 2);
    if (idx >= total) return;
    int  i  = (int)(idx & 63);          // pair index 0..63
    long th = idx >> 6;
    int  h  = (int)(th % heads);
    long t  = th / heads;               // global token
    float f = freqs[(t & (TSEQ - 1)) * (HD / 2) + i];
    float s, c;
    sincosf(f, &s, &c);
    long base = t * ((long)heads * HD) + (long)h * HD + 2 * i;
    float xr = __bfloat162float(buf[base]);
    float xi = __bfloat162float(buf[base + 1]);
    buf[base]     = __float2bfloat16((xr * c - xi * s) * oscale);
    buf[base + 1] = __float2bfloat16((xr * s + xi * c) * oscale);
}

// ---------------------------------------------------------------------------
// V transpose: v[token][kvh*HD + d] -> vt[((b*NKV + kvh)*HD + d)*TSEQ + tt]
// ---------------------------------------------------------------------------
__global__ void __launch_bounds__(256)
v_transpose(const __hip_bfloat16* __restrict__ v, __hip_bfloat16* __restrict__ vt) {
    long idx = (long)blockIdx.x * blockDim.x + threadIdx.x;
    long total = (long)TOK * NKV * HD;
    if (idx >= total) return;
    int  c  = (int)(idx & (NKV * HD - 1));  // column within row (1024)
    long r  = idx >> 10;                    // token
    int  b  = (int)(r >> 10);
    int  tt = (int)(r & (TSEQ - 1));
    int  kvh = c >> 7;
    int  d   = c & (HD - 1);
    vt[(((long)(b * NKV + kvh) * HD + d) * TSEQ) + tt] = v[idx];
}

// ---------------------------------------------------------------------------
// Flash attention: grid.x = B*NH (=128), grid.y = T/16 (=64), block = 1 wave.
// 64-key tiles; online softmax; row sums via ones-WMMA (no lane reduction);
// row max via ds_swizzle butterflies.
// ---------------------------------------------------------------------------
__global__ void __launch_bounds__(32, 1)
attn_kernel(const __hip_bfloat16* __restrict__ qbuf,
            const __hip_bfloat16* __restrict__ kbuf,
            const __hip_bfloat16* __restrict__ vt,
            __hip_bfloat16* __restrict__ obuf) {
    const int bh  = blockIdx.x;
    const int b   = bh >> 5;
    const int h   = bh & 31;
    const int kvh = h >> 2;                 // GQA: rep = 4
    const int qb  = blockIdx.y * 16;
    const int lane = threadIdx.x & 31;
    const int m_   = lane & 15;
    const int g    = lane >> 4;

    // Q fragments (scale already folded in by rope_kernel)
    const __hip_bfloat16* qrow =
        qbuf + ((size_t)(b * TSEQ + qb + m_)) * (NH * HD) + h * HD + 8 * g;
    v16bf qf[4];
#pragma unroll
    for (int i = 0; i < 4; ++i) qf[i] = load_frag16(qrow + i * 32);

    v8f   o[8] = {};
    v8f   lacc = {};                        // running row sums (every column equal)
    float mrow[8];
#pragma unroll
    for (int j = 0; j < 8; ++j) mrow[j] = -3.0e38f;

    const v16bf onesf = ones_frag();
    __shared__ __align__(16) __hip_bfloat16 pls[16 * 64];

    const __hip_bfloat16* kbase =
        kbuf + ((size_t)(b * TSEQ)) * (NKV * HD) + kvh * HD + 8 * g;
    const __hip_bfloat16* vbase =
        vt + ((size_t)(b * NKV + kvh) * HD + m_) * TSEQ + 8 * g;

    for (int kb = 0; kb < TSEQ; kb += 64) {
        // ---- scores: S[16 x 64] = Q (16x128) * K_tile^T ----
        v8f s[4] = {};
#pragma unroll
        for (int i = 0; i < 4; ++i) {
#pragma unroll
            for (int t = 0; t < 4; ++t) {
                v16bf kf = load_frag16(
                    kbase + (size_t)(kb + t * 16 + m_) * (NKV * HD) + i * 32);
                s[t] = wmma_bf16(qf[i], kf, s[t]);
            }
        }

        // ---- online softmax update (rows live across 16-lane groups) ----
        float alpha[8];
#pragma unroll
        for (int j = 0; j < 8; ++j) {
            float mx = fmaxf(fmaxf(s[0][j], s[1][j]), fmaxf(s[2][j], s[3][j]));
            mx = swz_max16(mx);
            float newm = fmaxf(mrow[j], mx);
            alpha[j] = __expf(mrow[j] - newm);
            mrow[j]  = newm;
            int r = j + 8 * g;                       // row in C-fragment layout
#pragma unroll
            for (int t = 0; t < 4; ++t) {
                float p = __expf(s[t][j] - newm);
                pls[r * 64 + t * 16 + m_] = __float2bfloat16(p);
            }
        }
        // rescale accumulators (and running sums) by alpha per row
#pragma unroll
        for (int nt = 0; nt < 8; ++nt)
#pragma unroll
            for (int j = 0; j < 8; ++j) o[nt][j] = o[nt][j] * alpha[j];
#pragma unroll
        for (int j = 0; j < 8; ++j) lacc[j] = lacc[j] * alpha[j];

        __syncthreads();
        // ---- P as two A-fragments (16x64, ld=64) ----
        v16bf pf0 = load_frag16(&pls[m_ * 64 + 8 * g]);
        v16bf pf1 = load_frag16(&pls[m_ * 64 + 32 + 8 * g]);

        // row sums of P via matrix pipe: l += P x ones
        lacc = wmma_bf16(pf0, onesf, lacc);
        lacc = wmma_bf16(pf1, onesf, lacc);

        // O += P x V  (V^T layout: contiguous along keys)
#pragma unroll
        for (int nt = 0; nt < 8; ++nt) {
            v16bf vf0 = load_frag16(vbase + (size_t)(nt * 16) * TSEQ + kb);
            v16bf vf1 = load_frag16(vbase + (size_t)(nt * 16) * TSEQ + kb + 32);
            o[nt] = wmma_bf16(pf0, vf0, o[nt]);
            o[nt] = wmma_bf16(pf1, vf1, o[nt]);
        }
        __syncthreads();
    }

    // ---- normalize and store to [token][NH*HD] bf16 ----
#pragma unroll
    for (int nt = 0; nt < 8; ++nt) {
#pragma unroll
        for (int j = 0; j < 8; ++j) {
            long r = (long)b * TSEQ + qb + j + 8 * g;
            int  c = h * HD + nt * 16 + m_;
            obuf[r * (NH * HD) + c] = __float2bfloat16(o[nt][j] / lacc[j]);
        }
    }
}

// ---------------------------------------------------------------------------
// Host-side orchestration
// ---------------------------------------------------------------------------
extern "C" void kernel_launch(void* const* d_in, const int* in_sizes, int n_in,
                              void* d_out, int out_size, void* d_ws, size_t ws_size,
                              hipStream_t stream) {
    const float* x     = (const float*)d_in[0];
    const float* freqs = (const float*)d_in[1];
    const float* wq    = (const float*)d_in[2];
    const float* wk    = (const float*)d_in[3];
    const float* wv    = (const float*)d_in[4];
    const float* wo    = (const float*)d_in[5];
    float* out = (float*)d_out;

    char* ws = (char*)d_ws;
    size_t off = 0;
    auto take = [&](size_t elems) {
        __hip_bfloat16* p = (__hip_bfloat16*)(ws + off);
        off += elems * sizeof(__hip_bfloat16);
        return p;
    };
    __hip_bfloat16* xb   = take((size_t)TOK * DMODEL);
    __hip_bfloat16* wqb  = take((size_t)NH * HD * DMODEL);
    __hip_bfloat16* wkb  = take((size_t)NKV * HD * DMODEL);
    __hip_bfloat16* wvb  = take((size_t)NKV * HD * DMODEL);
    __hip_bfloat16* wob  = take((size_t)DMODEL * NH * HD);
    __hip_bfloat16* qbuf = take((size_t)TOK * NH * HD);
    __hip_bfloat16* kbuf = take((size_t)TOK * NKV * HD);
    __hip_bfloat16* vbuf = take((size_t)TOK * NKV * HD);
    __hip_bfloat16* vtb  = take((size_t)TOK * NKV * HD);
    __hip_bfloat16* abuf = take((size_t)TOK * NH * HD);

    const int CT = 256;
    auto cvt = [&](const float* src, __hip_bfloat16* dst, long n) {
        cvt_f32_bf16<<<(unsigned)((n + CT - 1) / CT), CT, 0, stream>>>(src, dst, n);
    };
    cvt(x,  xb,  (long)TOK * DMODEL);
    cvt(wq, wqb, (long)NH * HD * DMODEL);
    cvt(wk, wkb, (long)NKV * HD * DMODEL);
    cvt(wv, wvb, (long)NKV * HD * DMODEL);
    cvt(wo, wob, (long)DMODEL * NH * HD);

    // QKV projections (bf16 out)
    {
        dim3 blk(128);
        dim3 gq(TOK / 128, (NH * HD) / 64);
        gemm_bf16_nt<__hip_bfloat16><<<gq, blk, 0, stream>>>(xb, wqb, qbuf,
                                                             TOK, NH * HD, DMODEL);
        dim3 gkv(TOK / 128, (NKV * HD) / 64);
        gemm_bf16_nt<__hip_bfloat16><<<gkv, blk, 0, stream>>>(xb, wkb, kbuf,
                                                              TOK, NKV * HD, DMODEL);
        gemm_bf16_nt<__hip_bfloat16><<<gkv, blk, 0, stream>>>(xb, wvb, vbuf,
                                                              TOK, NKV * HD, DMODEL);
    }

    // RoPE on Q (with 1/sqrt(HD) folded in) and K
    {
        long nq = (long)TOK * NH * (HD / 2);
        long nk = (long)TOK * NKV * (HD / 2);
        rope_kernel<<<(unsigned)((nq + CT - 1) / CT), CT, 0, stream>>>(
            qbuf, freqs, NH, 0.08838834764831845f);
        rope_kernel<<<(unsigned)((nk + CT - 1) / CT), CT, 0, stream>>>(
            kbuf, freqs, NKV, 1.0f);
    }

    // Transpose V for contiguous P·V fragment loads
    {
        long nv = (long)TOK * NKV * HD;
        v_transpose<<<(unsigned)((nv + CT - 1) / CT), CT, 0, stream>>>(vbuf, vtb);
    }

    // Flash attention
    {
        dim3 grid(4 * NH, TSEQ / 16);
        attn_kernel<<<grid, 32, 0, stream>>>(qbuf, kbuf, vtb, abuf);
    }

    // Output projection (fp32 out)
    {
        dim3 blk(128);
        dim3 go(TOK / 128, DMODEL / 64);
        gemm_bf16_nt<float><<<go, blk, 0, stream>>>(abuf, wob, out,
                                                    TOK, DMODEL, NH * HD);
    }
}
